// DHDN_Bigger_25065429139912
// MI455X (gfx1250) — matrix operand order repaired and checked by reading, not compiled, not run on hardware
//
#include <hip/hip_runtime.h>
#include <hip/hip_bf16.h>

// ---------------------------------------------------------------------------
// MI455X (gfx1250) implementation.
//   * All convs + dense layers -> one implicit-GEMM kernel using
//     v_wmma_f32_16x16x32_f16 (f16 inputs, fp32 accumulate), 128x64x32 tiles,
//     8 waves/WG (wave32), LDS-staged fragments in the CDNA5 layouts.
//   * Activations carried in f16 alongside f32 so A/B tiles are raw byte
//     copies -> staged with GLOBAL_LOAD_ASYNC_TO_LDS_B128 (ASYNCcnt) where
//     contiguous; im2col gather path for 7x7/3x3/strided convs.
//   * Small tail ops (LN, MHA softmax, hypergraph conv, knn, VAE) are VALU.
// ---------------------------------------------------------------------------

typedef __attribute__((ext_vector_type(16))) _Float16 v16h;
typedef __attribute__((ext_vector_type(8)))  _Float16 v8h;
typedef __attribute__((ext_vector_type(8)))  float    v8f;

#define BM 128
#define BN 64
#define BK 32

#define NB  2
#define NJ  19
#define NIMG 38
#define HID 128

#define ASYNC_COPY 1

// ------------------------ async copy helpers (CDNA5) -----------------------
__device__ __forceinline__ void async_ld_b128(const _Float16* lds_dst, const _Float16* gsrc) {
  unsigned loff = (unsigned)(size_t)lds_dst;   // low 32 bits of generic ptr == LDS offset
  asm volatile("global_load_async_to_lds_b128 %0, %1, off"
               :: "v"(loff), "v"(gsrc) : "memory");
}
__device__ __forceinline__ void async_wait_all() {
  asm volatile("s_wait_asynccnt 0x0" ::: "memory");
}
__device__ __forceinline__ v8h zero8h() {
  v8h z;
  #pragma unroll
  for (int t = 0; t < 8; ++t) z[t] = (_Float16)0.f;
  return z;
}

// --------------------------- utility kernels -------------------------------
__global__ __launch_bounds__(256)
void f32_to_f16_kernel(const float* __restrict__ s, _Float16* __restrict__ d, long n) {
  long i = (long)blockIdx.x * 256 + threadIdx.x;
  if (i < n) d[i] = (_Float16)s[i];
}

// W [K,N] fp32  ->  Wt [N,Kp] f16 (zero-padded K -> Kp)
__global__ __launch_bounds__(256)
void f32_to_f16_T_kernel(const float* __restrict__ src, _Float16* __restrict__ dst,
                         int K, int N, int Kp) {
  long i = (long)blockIdx.x * 256 + threadIdx.x;
  long total = (long)N * Kp;
  if (i >= total) return;
  int n = (int)(i / Kp), k = (int)(i % Kp);
  dst[i] = (k < K) ? (_Float16)src[(long)k * N + n] : (_Float16)0.f;
}

// ------------------- WMMA implicit-GEMM conv / linear ----------------------
// Out[M,N] = im2col(A16) @ Wt^T, A16 NHWC f16, Wt [N,Kp] f16 (pre-transposed).
// FAST: KH==KW==1, stride==1, pad==0, K%32==0 -> contiguous async A staging.
// Epilogue: +bias[n]; (bn if !bnAfter); +residual; relu; (bn if bnAfter)
template <bool FAST>
__global__ __launch_bounds__(256)
void conv_gemm_wmma(const _Float16* __restrict__ A16,
                    const _Float16* __restrict__ Wt,
                    float* __restrict__ Out, _Float16* __restrict__ Out16,
                    int nImg, int IH, int IW, int Cin,
                    int KH, int KW, int stride, int pad,
                    int OH, int OW, int Cout, int Kp,
                    const float* __restrict__ bias,
                    const float* __restrict__ bn_g,
                    const float* __restrict__ bn_b,
                    const float* __restrict__ residual,
                    int relu_flag, int bn_after_relu)
{
  const int M = nImg * OH * OW;
  const int N = Cout;
  const int K = KH * KW * Cin;
  const int KWC = KW * Cin;

  __shared__ _Float16 As[BM * BK];   // [m][k]
  __shared__ _Float16 Bs[BN * BK];   // [n][k]

  const int tid  = threadIdx.x;
  const int lane = tid & 31;
  const int wave = tid >> 5;
  const int wm   = wave >> 1;
  const int wn   = wave & 1;
  const int m0   = blockIdx.x * BM;
  const int n0   = blockIdx.y * BN;

  // per-thread staging coordinates (K-invariant, hoisted)
  const int arow = tid >> 1;            // 0..127
  const int akb  = (tid & 1) * 16;      // 0 or 16
  const int am   = m0 + arow;
  const bool amok = am < M;
  int a_nimg = 0, a_oh = 0, a_ow = 0;
  if (!FAST && amok) {
    a_nimg = am / (OH * OW);
    int r = am - a_nimg * OH * OW;
    a_oh = r / OW; a_ow = r - a_oh * OW;
  }
  const _Float16* Arow = A16 + (long)am * K;          // FAST path only

  const int brow = tid >> 2;            // 0..63
  const int bkc  = (tid & 3) * 8;       // 0,8,16,24
  const bool bok = (n0 + brow) < N;
  const _Float16* Bp = Wt + (long)(n0 + brow) * Kp + bkc;

  v8f acc[2][2] = {};

  for (int k0 = 0; k0 < K; k0 += BK) {
    // ---- stage B tile (contiguous; Kp zero-padded so no K guard) ----
    if (bok) {
#if ASYNC_COPY
      async_ld_b128(&Bs[brow * BK + bkc], Bp + k0);
#else
      *(v8h*)&Bs[brow * BK + bkc] = *(const v8h*)(Bp + k0);
#endif
    } else {
      *(v8h*)&Bs[brow * BK + bkc] = zero8h();
    }
    // ---- stage A tile ----
    if (FAST) {
      if (amok) {
#if ASYNC_COPY
        async_ld_b128(&As[arow * BK + akb],     Arow + k0 + akb);
        async_ld_b128(&As[arow * BK + akb + 8], Arow + k0 + akb + 8);
#else
        *(v8h*)&As[arow * BK + akb]       = *(const v8h*)(Arow + k0 + akb);
        *(v8h*)&As[arow * BK + akb + 8]   = *(const v8h*)(Arow + k0 + akb + 8);
#endif
      } else {
        *(v8h*)&As[arow * BK + akb]     = zero8h();
        *(v8h*)&As[arow * BK + akb + 8] = zero8h();
      }
    } else {
      _Float16* dst = &As[arow * BK + akb];
      #pragma unroll
      for (int t = 0; t < 16; ++t) {
        const int k = k0 + akb + t;
        _Float16 v = (_Float16)0.f;
        if (amok && k < K) {
          const int kh = k / KWC;
          const int r2 = k - kh * KWC;
          const int kw = r2 / Cin;
          const int ci = r2 - kw * Cin;
          const int ih = a_oh * stride + kh - pad;
          const int iw = a_ow * stride + kw - pad;
          if (ih >= 0 && ih < IH && iw >= 0 && iw < IW)
            v = A16[(((long)a_nimg * IH + ih) * IW + iw) * Cin + ci];
        }
        dst[t] = v;
      }
    }
#if ASYNC_COPY
    async_wait_all();
#endif
    __syncthreads();

    // ---- fragments per CDNA5 wave32 layouts ----
    v16h afrag[2], bfrag[2];
    #pragma unroll
    for (int i = 0; i < 2; ++i) {
      const _Float16* ar = &As[(wm * 32 + i * 16 + (lane & 15)) * BK];
      const int ka = (lane < 16) ? 0 : 8;
      #pragma unroll
      for (int t = 0; t < 8; ++t) {
        afrag[i][t]     = ar[ka + t];
        afrag[i][t + 8] = ar[ka + 16 + t];
      }
    }
    #pragma unroll
    for (int j = 0; j < 2; ++j) {
      const _Float16* br = &Bs[(wn * 32 + j * 16 + (lane & 15)) * BK];
      const int kb2 = (lane < 16) ? 0 : 8;
      #pragma unroll
      for (int t = 0; t < 8; ++t) {
        bfrag[j][t]     = br[kb2 + t];
        bfrag[j][t + 8] = br[kb2 + 16 + t];
      }
    }

    #pragma unroll
    for (int i = 0; i < 2; ++i)
      #pragma unroll
      for (int j = 0; j < 2; ++j)
        acc[i][j] = __builtin_amdgcn_wmma_f32_16x16x32_f16(
            false, afrag[i], false, bfrag[j], (short)0, acc[i][j], false, false);
    __syncthreads();
  }

  // ---- epilogue + store ----
  const float BN_MUL = 0.99999500003749969f;  // 1/sqrt(1+1e-5)
  #pragma unroll
  for (int i = 0; i < 2; ++i) {
    #pragma unroll
    for (int j = 0; j < 2; ++j) {
      #pragma unroll
      for (int r = 0; r < 8; ++r) {
        const int row = wm * 32 + i * 16 + r + ((lane < 16) ? 0 : 8);
        const int col = wn * 32 + j * 16 + (lane & 15);
        const int m = m0 + row, n = n0 + col;
        if (m < M && n < N) {
          float v = acc[i][j][r];
          if (bias) v += bias[n];
          if (bn_g && !bn_after_relu) v = v * (bn_g[n] * BN_MUL) + bn_b[n];
          if (residual) v += residual[(long)m * N + n];
          if (relu_flag) v = v > 0.f ? v : 0.f;
          if (bn_g && bn_after_relu) v = v * (bn_g[n] * BN_MUL) + bn_b[n];
          if (Out)   Out[(long)m * N + n] = v;
          if (Out16) Out16[(long)m * N + n] = (_Float16)v;
        }
      }
    }
  }
}

// ------------------------- patch extraction --------------------------------
__global__ __launch_bounds__(224)
void extract_patches_kernel(const float* __restrict__ img,   // [B,3,256,256]
                            const float* __restrict__ uv,    // [B, 2J]
                            _Float16* __restrict__ patches)  // [38,224,224,3] f16
{
  const int ox = threadIdx.x, oy = blockIdx.x, p = blockIdx.y;
  const int b = p / NJ, j = p % NJ;
  const float px = uv[b * (2 * NJ) + j * 2 + 0];
  const float py = uv[b * (2 * NJ) + j * 2 + 1];
  int xi = (int)floorf((px + 0.5f) * 256.f);
  int yi = (int)floorf((py + 0.5f) * 256.f);
  int xlo = max(0, xi - 8), xhi = min(256, xi + 8);
  if (xhi <= xlo) xlo = xhi - 1;
  int ylo = max(0, yi - 8), yhi = min(256, yi + 8);
  if (yhi <= ylo) ylo = yhi - 1;

  const float nx = (float)(xhi - xlo);
  float sx = (ox + 0.5f) * (nx / 224.f) - 0.5f;
  sx = fminf(fmaxf(sx, 0.f), fmaxf(nx - 1.f, 0.f));
  int ix0 = (int)floorf(sx);
  const float wx = sx - (float)ix0;
  int x0 = xlo + ix0;
  int x1 = xlo + min(ix0 + 1, xhi - xlo - 1);

  const float ny = (float)(yhi - ylo);
  float sy = (oy + 0.5f) * (ny / 224.f) - 0.5f;
  sy = fminf(fmaxf(sy, 0.f), fmaxf(ny - 1.f, 0.f));
  int iy0 = (int)floorf(sy);
  const float wy = sy - (float)iy0;
  int y0 = ylo + iy0;
  int y1 = ylo + min(iy0 + 1, yhi - ylo - 1);

  #pragma unroll
  for (int ch = 0; ch < 3; ++ch) {
    const float* pl = &img[((long)(b * 3 + ch)) * 256 * 256];
    const float g00 = pl[y0 * 256 + x0], g01 = pl[y0 * 256 + x1];
    const float g10 = pl[y1 * 256 + x0], g11 = pl[y1 * 256 + x1];
    const float top = (1.f - wx) * g00 + wx * g01;
    const float bot = (1.f - wx) * g10 + wx * g11;
    patches[(((long)p * 224 + oy) * 224 + ox) * 3 + ch] =
        (_Float16)((1.f - wy) * top + wy * bot);
  }
}

// ----------------------------- pooling -------------------------------------
__global__ __launch_bounds__(256)
void maxpool_kernel(const float* __restrict__ in, _Float16* __restrict__ out16) {
  long i = (long)blockIdx.x * 256 + threadIdx.x;
  const long total = (long)NIMG * 56 * 56 * 64;
  if (i >= total) return;
  int c = (int)(i & 63); long r = i >> 6;
  int ow = (int)(r % 56); r /= 56;
  int oh = (int)(r % 56); int n = (int)(r / 56);
  float m = -3.4e38f;
  #pragma unroll
  for (int dy = 0; dy < 3; ++dy)
    #pragma unroll
    for (int dx = 0; dx < 3; ++dx) {
      int ih = oh * 2 - 1 + dy, iw = ow * 2 - 1 + dx;
      if (ih >= 0 && ih < 112 && iw >= 0 && iw < 112)
        m = fmaxf(m, in[(((long)n * 112 + ih) * 112 + iw) * 64 + c]);
    }
  out16[i] = (_Float16)m;
}

__global__ __launch_bounds__(256)
void avgpool_kernel(const float* __restrict__ in, _Float16* __restrict__ out16,
                    int HW, int C) {
  int i = blockIdx.x * 256 + threadIdx.x;
  if (i >= NIMG * C) return;
  int n = i / C, c = i % C;
  float s = 0.f;
  for (int t = 0; t < HW; ++t) s += in[((long)n * HW + t) * C + c];
  out16[i] = (_Float16)(s / (float)HW);
}

// ------------------------------ layernorm ----------------------------------
__global__ __launch_bounds__(HID)
void ln_kernel(const float* __restrict__ x, const float* __restrict__ addend,
               const float* __restrict__ g, const float* __restrict__ b,
               float* __restrict__ out, _Float16* __restrict__ out16, int prerelu)
{
  __shared__ float sm[HID];
  const int row = blockIdx.x, c = threadIdx.x;
  float v = x[row * HID + c];
  if (addend) v += addend[row * HID + c];
  if (prerelu) v = fmaxf(v, 0.f);
  sm[c] = v; __syncthreads();
  for (int s = HID / 2; s > 0; s >>= 1) { if (c < s) sm[c] += sm[c + s]; __syncthreads(); }
  const float mean = sm[0] * (1.f / HID);
  __syncthreads();
  const float d = v - mean;
  sm[c] = d * d; __syncthreads();
  for (int s = HID / 2; s > 0; s >>= 1) { if (c < s) sm[c] += sm[c + s]; __syncthreads(); }
  const float var = sm[0] * (1.f / HID);
  const float r = d * rsqrtf(var + 1e-5f) * g[c] + b[c];
  out[row * HID + c] = r;
  if (out16) out16[row * HID + c] = (_Float16)r;
}

__global__ __launch_bounds__(256)
void add_kernel(const float* a, const float* b, float* o, _Float16* o16, int n) {
  int i = blockIdx.x * 256 + threadIdx.x;
  if (i < n) { float v = a[i] + b[i]; o[i] = v; o16[i] = (_Float16)v; }
}

// ---------------------------- knn / topk -----------------------------------
__global__ __launch_bounds__(64)
void knn_kernel(const float* __restrict__ uv, const float* __restrict__ imgf,
                int* __restrict__ edge)
{
  int i = threadIdx.x; if (i >= NIMG) return;
  int b = i / NJ, j = i % NJ;
  float comb[NJ];
  const float px = uv[b * 2 * NJ + j * 2], py = uv[b * 2 * NJ + j * 2 + 1];
  const float* fa = &imgf[(b * NJ + j) * HID];
  for (int jj = 0; jj < NJ; ++jj) {
    float dx = px - uv[b * 2 * NJ + jj * 2];
    float dy = py - uv[b * 2 * NJ + jj * 2 + 1];
    float dp = sqrtf(fmaxf(dx * dx + dy * dy, 0.f));
    const float* fb = &imgf[(b * NJ + jj) * HID];
    float s = 0.f;
    for (int c = 0; c < HID; ++c) { float d = fa[c] - fb[c]; s += d * d; }
    comb[jj] = 0.5f * dp + 0.5f * sqrtf(fmaxf(s, 0.f));
  }
  bool used[NJ];
  for (int t = 0; t < NJ; ++t) used[t] = false;
  for (int k = 0; k < 5; ++k) {
    int am = 0; float bv = 3.4e38f;
    for (int jj = 0; jj < NJ; ++jj)
      if (!used[jj] && comb[jj] < bv) { bv = comb[jj]; am = jj; }
    used[am] = true;
    edge[i * 5 + k] = b * NJ + am;
  }
}

// --------------------------- hypergraph conv -------------------------------
__global__ __launch_bounds__(HID)
void hgnn_edge_kernel(const float* __restrict__ xt, const int* __restrict__ edge,
                      float* __restrict__ efeat)
{
  const int e = blockIdx.x, c = threadIdx.x;
  float s = 0.f; int cnt = 0;
  for (int i = 0; i < NIMG * 5; ++i)
    if (edge[i] == e) { s += xt[(i / 5) * HID + c]; ++cnt; }
  efeat[e * HID + c] = cnt > 0 ? s / (float)cnt : 0.f;
}

__global__ __launch_bounds__(HID)
void hgnn_node_kernel(const float* __restrict__ efeat, const int* __restrict__ edge,
                      const float* __restrict__ bias, float* __restrict__ out)
{
  const int nId = blockIdx.x, c = threadIdx.x;
  float s = 0.f;
  #pragma unroll
  for (int k = 0; k < 5; ++k) s += efeat[edge[nId * 5 + k] * HID + c];
  out[nId * HID + c] = s * 0.2f + bias[c];
}

// ----------------------------- attention -----------------------------------
__global__ __launch_bounds__(32)
void mha_kernel(const float* __restrict__ qkv, _Float16* __restrict__ o16) {
  const int b = blockIdx.x >> 3, h = blockIdx.x & 7;
  const int q = threadIdx.x;
  if (q >= NJ) return;
  const float* qp = &qkv[((long)(b * NJ + q)) * 384 + h * 16];
  float sc[NJ]; float mx = -3.4e38f;
  for (int kk = 0; kk < NJ; ++kk) {
    const float* kp = &qkv[((long)(b * NJ + kk)) * 384 + 128 + h * 16];
    float d = 0.f;
    #pragma unroll
    for (int t = 0; t < 16; ++t) d += qp[t] * kp[t];
    d *= 0.25f;
    sc[kk] = d; mx = fmaxf(mx, d);
  }
  float se = 0.f;
  for (int kk = 0; kk < NJ; ++kk) { sc[kk] = __expf(sc[kk] - mx); se += sc[kk]; }
  const float inv = 1.f / se;
  #pragma unroll
  for (int t = 0; t < 16; ++t) {
    float acc = 0.f;
    for (int kk = 0; kk < NJ; ++kk)
      acc += sc[kk] * qkv[((long)(b * NJ + kk)) * 384 + 256 + h * 16 + t];
    o16[(long)(b * NJ + q) * HID + h * 16 + t] = (_Float16)(acc * inv);
  }
}

// -------------------------------- VAE --------------------------------------
__global__ __launch_bounds__(64)
void vae_kernel(const float* __restrict__ zp, const float* __restrict__ eps,
                float* __restrict__ out_mu, float* __restrict__ out_lv,
                float* __restrict__ out_z, _Float16* __restrict__ z16)
{
  int i = threadIdx.x; if (i >= 40) return;
  int b = i / 20, r = i % 20, p = r / 2, d = r % 2;
  float mu = zp[b * 40 + p * 4 + d];
  float lv = zp[b * 40 + p * 4 + 2 + d];
  float z  = mu + eps[i] * __expf(0.5f * lv);
  out_mu[i] = mu; out_lv[i] = lv; out_z[i] = z; z16[i] = (_Float16)z;
}

// ============================ host orchestration ============================

extern "C" void kernel_launch(void* const* d_in, const int* in_sizes, int n_in,
                              void* d_out, int out_size, void* d_ws, size_t ws_size,
                              hipStream_t stream) {
  (void)in_sizes; (void)n_in; (void)out_size; (void)ws_size;

  const float* d_img = (const float*)d_in[0];
  const float* d_uv  = (const float*)d_in[1];
  const float* d_eps = (const float*)d_in[2];

  // ---- param walk (JAX pytree order: dict keys sorted alphabetically) ----
  int pi = 3;
  auto nx = [&]() -> const float* { return (const float*)d_in[pi++]; };

  struct MlpL { const float *W, *b, *beta, *g; };
  MlpL dec[3], enc[3], refl[3];
  for (int i = 0; i < 3; ++i) {
    dec[i].W = nx(); dec[i].b = nx();
    dec[i].beta = (i < 2) ? nx() : nullptr;
    dec[i].g    = (i < 2) ? nx() : nullptr;
  }
  for (int i = 0; i < 3; ++i) {
    enc[i].W = nx(); enc[i].b = nx();
    enc[i].beta = (i < 2) ? nx() : nullptr;
    enc[i].g    = (i < 2) ? nx() : nullptr;
  }
  const float* fpn_b    = nx();
  const float* fpn_beta = nx();
  const float* fpn_g    = nx();
  const float* fpn_w    = nx();
  struct HgL { const float *beta, *bias, *g, *theta; } hg[3];
  for (int i = 0; i < 3; ++i) { hg[i].beta = nx(); hg[i].bias = nx(); hg[i].g = nx(); hg[i].theta = nx(); }
  const float* hi_b    = nx();
  const float* hi_beta = nx();
  const float* hi_g    = nx();
  const float* hi_w    = nx();
  const float* proj_b  = nx();
  const float* proj_w  = nx();
  for (int i = 0; i < 3; ++i) {
    refl[i].W = nx(); refl[i].b = nx();
    refl[i].beta = (i < 2) ? nx() : nullptr;
    refl[i].g    = (i < 2) ? nx() : nullptr;
  }
  const float* rn_b1    = nx();
  const float* rn_conv1 = nx();
  const float* rn_g1    = nx();
  struct BlockP { const float *b1,*b2,*b3,*bd,*g1,*g2,*g3,*gd,*w1,*w2,*w3,*wd; int down; };
  BlockP blk[16];
  const int nb[4] = {3, 4, 6, 3};
  {
    int bi = 0;
    for (int s = 0; s < 4; ++s)
      for (int k = 0; k < nb[s]; ++k, ++bi) {
        BlockP& p = blk[bi]; p.down = (k == 0);
        p.b1 = nx(); p.b2 = nx(); p.b3 = nx(); p.bd = p.down ? nx() : nullptr;
        p.g1 = nx(); p.g2 = nx(); p.g3 = nx(); p.gd = p.down ? nx() : nullptr;
        p.w1 = nx(); p.w2 = nx(); p.w3 = nx(); p.wd = p.down ? nx() : nullptr;
      }
  }
  struct TfL { const float *Wf1,*Wf2,*Wo,*Wqkv,*b1,*b2,*bf1,*bf2,*bo,*bqkv,*g1,*g2; } tf[2];
  for (int i = 0; i < 2; ++i) {
    tf[i].Wf1 = nx(); tf[i].Wf2 = nx(); tf[i].Wo = nx(); tf[i].Wqkv = nx();
    tf[i].b1 = nx(); tf[i].b2 = nx(); tf[i].bf1 = nx(); tf[i].bf2 = nx();
    tf[i].bo = nx(); tf[i].bqkv = nx(); tf[i].g1 = nx(); tf[i].g2 = nx();
  }

  // ---- workspace allocator ----
  size_t wofs = 0;
  auto alloc = [&](size_t bytes) -> void* {
    wofs = (wofs + 255) & ~(size_t)255;
    void* p = (char*)d_ws + wofs;
    wofs += bytes;
    return p;
  };
  // transposed + K-padded f16 weights
  auto cvtT = [&](const float* src, int K, int N) -> const _Float16* {
    const int Kp = (K + 31) & ~31;
    const long tot = (long)N * Kp;
    _Float16* p = (_Float16*)alloc((size_t)tot * 2);
    f32_to_f16_T_kernel<<<(unsigned)((tot + 255) / 256), 256, 0, stream>>>(src, p, K, N, Kp);
    return p;
  };

  // ---- weight conversion ----
  const _Float16* conv1T = cvtT(rn_conv1, 147, 64);
  const _Float16 *w1T[16], *w2T[16], *w3T[16], *wdT[16];
  {
    int cin = 64; const int widths[4] = {64, 128, 256, 512}; int bi = 0;
    for (int s = 0; s < 4; ++s) {
      const int w = widths[s];
      for (int k = 0; k < nb[s]; ++k, ++bi) {
        w1T[bi] = cvtT(blk[bi].w1, cin, w);
        w2T[bi] = cvtT(blk[bi].w2, 9 * w, w);
        w3T[bi] = cvtT(blk[bi].w3, w, 4 * w);
        wdT[bi] = blk[bi].down ? cvtT(blk[bi].wd, cin, 4 * w) : nullptr;
        cin = 4 * w;
      }
    }
  }
  const _Float16* fpnT  = cvtT(fpn_w, 2048, HID);
  const _Float16* projT = cvtT(proj_w, HID, HID);
  const _Float16* hiT   = cvtT(hi_w, 2, HID);
  const _Float16* hgT[3];
  for (int i = 0; i < 3; ++i) hgT[i] = cvtT(hg[i].theta, HID, HID);
  const _Float16 *qkvT[2], *woT[2], *wf1T[2], *wf2T[2];
  for (int i = 0; i < 2; ++i) {
    qkvT[i] = cvtT(tf[i].Wqkv, HID, 384);
    woT[i]  = cvtT(tf[i].Wo,  HID, HID);
    wf1T[i] = cvtT(tf[i].Wf1, HID, 512);
    wf2T[i] = cvtT(tf[i].Wf2, 512, HID);
  }
  const int encK[3] = {NJ * HID, 512, 256}, encN[3] = {512, 256, 40};
  const int decK[3] = {20, 256, 512},       decN[3] = {256, 512, NJ * HID};
  const int refK[3] = {NJ * HID, 512, 256}, refN[3] = {512, 256, NJ * 2};
  const _Float16 *encT[3], *decT[3], *refT[3];
  for (int i = 0; i < 3; ++i) {
    encT[i] = cvtT(enc[i].W, encK[i], encN[i]);
    decT[i] = cvtT(dec[i].W, decK[i], decN[i]);
    refT[i] = cvtT(refl[i].W, refK[i], refN[i]);
  }

  // ---- activation buffers ----
  const size_t BIG = (size_t)NIMG * 112 * 112 * 64 * 4;   // 122 MB fp32
  _Float16* patches16 = (_Float16*)alloc((size_t)NIMG * 224 * 224 * 3 * 2);
  float*    bufA  = (float*)alloc(BIG);
  float*    bufB  = (float*)alloc(BIG);
  _Float16* bufAh = (_Float16*)alloc(BIG / 2);
  _Float16* bufBh = (_Float16*)alloc(BIG / 2);
  float*    bufDS = (float*)alloc(BIG);
  _Float16* pool16 = (_Float16*)alloc((size_t)NIMG * 56 * 56 * 64 * 2);
  _Float16* t1h = (_Float16*)alloc((size_t)NIMG * 56 * 56 * 128 * 2);
  _Float16* t2h = (_Float16*)alloc((size_t)NIMG * 56 * 56 * 128 * 2);
  _Float16* feats16 = (_Float16*)alloc((size_t)NIMG * 2048 * 2);
  _Float16* f1h     = (_Float16*)alloc((size_t)NIMG * HID * 2);
  float*    imgfeat = (float*)alloc((size_t)NIMG * HID * 4);
  _Float16* uv16    = (_Float16*)alloc(NB * 2 * NJ * 2);
  float*    hln     = (float*)alloc((size_t)NIMG * HID * 4);
  float*    hx32[2]; _Float16* hx16[2];
  for (int i = 0; i < 2; ++i) {
    hx32[i] = (float*)alloc((size_t)NIMG * HID * 4);
    hx16[i] = (_Float16*)alloc((size_t)NIMG * HID * 2);
  }
  float*    xtb   = (float*)alloc((size_t)NIMG * HID * 4);
  float*    efeat = (float*)alloc((size_t)NIMG * HID * 4);
  int*      edges = (int*)alloc((size_t)NIMG * 5 * 4);
  float*    qkvbuf = (float*)alloc((size_t)NIMG * 384 * 4);
  _Float16* o16    = (_Float16*)alloc((size_t)NIMG * HID * 2);
  float*    tbuf   = (float*)alloc((size_t)NIMG * HID * 4);
  _Float16* ffh    = (_Float16*)alloc((size_t)NIMG * 512 * 2);
  _Float16* m0h    = (_Float16*)alloc(2 * 512 * 2);
  _Float16* m1h    = (_Float16*)alloc(2 * 512 * 2);
  float*    zp     = (float*)alloc(2 * 40 * 4);
  _Float16* zws16  = (_Float16*)alloc(40 * 2);
  _Float16* hrec16 = (_Float16*)alloc(2L * NJ * HID * 2);

  auto gemm = [&](const _Float16* A16, const _Float16* Wt, float* Outf, _Float16* Out16,
                  int nImg, int IH, int IW, int Cin, int KH, int KW, int st, int pad,
                  int OH, int OW, int Cout,
                  const float* bias, const float* g, const float* b,
                  const float* res, int relu, int bnAfter) {
    const int K = KH * KW * Cin;
    const int Kp = (K + 31) & ~31;
    const int M = nImg * OH * OW;
    dim3 grid((M + BM - 1) / BM, (Cout + BN - 1) / BN);
    const bool fast = (KH == 1 && KW == 1 && st == 1 && pad == 0 && (K % 32) == 0);
    if (fast)
      conv_gemm_wmma<true><<<grid, 256, 0, stream>>>(A16, Wt, Outf, Out16, nImg, IH, IW, Cin,
          KH, KW, st, pad, OH, OW, Cout, Kp, bias, g, b, res, relu, bnAfter);
    else
      conv_gemm_wmma<false><<<grid, 256, 0, stream>>>(A16, Wt, Outf, Out16, nImg, IH, IW, Cin,
          KH, KW, st, pad, OH, OW, Cout, Kp, bias, g, b, res, relu, bnAfter);
  };
  auto linear = [&](const _Float16* A16, const _Float16* Wt, float* Outf, _Float16* Out16,
                    int M, int Kd, int N,
                    const float* bias, const float* g, const float* b, int relu, int bnAfter) {
    gemm(A16, Wt, Outf, Out16, M, 1, 1, Kd, 1, 1, 1, 0, 1, 1, N, bias, g, b, nullptr, relu, bnAfter);
  };

  // ---- 1) patches ----
  {
    dim3 grid(224, NIMG);
    extract_patches_kernel<<<grid, 224, 0, stream>>>(d_img, d_uv, patches16);
  }
  // ---- 2) stem ----
  gemm(patches16, conv1T, bufA, nullptr, NIMG, 224, 224, 3, 7, 7, 2, 3, 112, 112, 64,
       nullptr, rn_g1, rn_b1, nullptr, 1, 0);
  {
    long total = (long)NIMG * 56 * 56 * 64;
    maxpool_kernel<<<(unsigned)((total + 255) / 256), 256, 0, stream>>>(bufA, pool16);
  }
  // ---- 3) bottleneck stages ----
  int finalHc = 7;
  const float* finalX32 = nullptr;
  {
    float*    xout32[2] = {bufA, bufB};
    _Float16* xout16[2] = {bufAh, bufBh};
    const _Float16* x16 = pool16;
    const float*    x32 = nullptr;
    int cur = 0;
    int cin = 64, Hc = 56;
    const int widths[4] = {64, 128, 256, 512};
    const int strides[4] = {1, 2, 2, 2};
    int bi = 0;
    for (int s = 0; s < 4; ++s) {
      const int w = widths[s];
      for (int k = 0; k < nb[s]; ++k, ++bi) {
        const BlockP& p = blk[bi];
        const int st = (k == 0) ? strides[s] : 1;
        const int OH = (Hc + 2 - 3) / st + 1;
        gemm(x16, w1T[bi], nullptr, t1h, NIMG, Hc, Hc, cin, 1, 1, 1, 0, Hc, Hc, w,
             nullptr, p.g1, p.b1, nullptr, 1, 0);
        gemm(t1h, w2T[bi], nullptr, t2h, NIMG, Hc, Hc, w, 3, 3, st, 1, OH, OH, w,
             nullptr, p.g2, p.b2, nullptr, 1, 0);
        const float* resid = x32;
        if (p.down) {
          gemm(x16, wdT[bi], bufDS, nullptr, NIMG, Hc, Hc, cin, 1, 1, st, 0, OH, OH, 4 * w,
               nullptr, p.gd, p.bd, nullptr, 0, 0);
          resid = bufDS;
        }
        gemm(t2h, w3T[bi], xout32[cur], xout16[cur], NIMG, OH, OH, w, 1, 1, 1, 0, OH, OH, 4 * w,
             nullptr, p.g3, p.b3, resid, 1, 0);
        x32 = xout32[cur]; x16 = xout16[cur]; cur ^= 1;
        cin = 4 * w; Hc = OH;
      }
    }
    finalHc = Hc; finalX32 = x32;
    // ---- 4) head ----
    avgpool_kernel<<<(NIMG * 2048 + 255) / 256, 256, 0, stream>>>(finalX32, feats16,
                                                                  finalHc * finalHc, 2048);
  }
  linear(feats16, fpnT, nullptr, f1h, NIMG, 2048, HID, fpn_b, fpn_g, fpn_beta, 1, 0);
  linear(f1h, projT, imgfeat, nullptr, NIMG, HID, HID, proj_b, nullptr, nullptr, 0, 0);
  f32_to_f16_kernel<<<1, 256, 0, stream>>>(d_uv, uv16, NB * 2 * NJ);
  linear(uv16, hiT, hln, nullptr, NIMG, 2, HID, hi_b, nullptr, nullptr, 1, 0);
  ln_kernel<<<NIMG, HID, 0, stream>>>(hln, nullptr, hi_g, hi_beta, hln, nullptr, 0);
  add_kernel<<<(NIMG * HID + 255) / 256, 256, 0, stream>>>(hln, imgfeat, hx32[0], hx16[0],
                                                           NIMG * HID);
  knn_kernel<<<1, 64, 0, stream>>>(d_uv, imgfeat, edges);
  // ---- 5) hypergraph conv x3 ----
  float* cx32 = hx32[0]; _Float16* cx16 = hx16[0];
  float* ox32 = hx32[1]; _Float16* ox16 = hx16[1];
  for (int i = 0; i < 3; ++i) {
    linear(cx16, hgT[i], xtb, nullptr, NIMG, HID, HID, nullptr, nullptr, nullptr, 0, 0);
    hgnn_edge_kernel<<<NIMG, HID, 0, stream>>>(xtb, edges, efeat);
    hgnn_node_kernel<<<NIMG, HID, 0, stream>>>(efeat, edges, hg[i].bias, ox32);
    ln_kernel<<<NIMG, HID, 0, stream>>>(ox32, nullptr, hg[i].g, hg[i].beta, ox32, ox16, 1);
    { float* t = cx32; cx32 = ox32; ox32 = t; }
    { _Float16* t = cx16; cx16 = ox16; ox16 = t; }
  }
  // ---- 6) transformer x2 ----
  for (int i = 0; i < 2; ++i) {
    linear(cx16, qkvT[i], qkvbuf, nullptr, NIMG, HID, 384, tf[i].bqkv, nullptr, nullptr, 0, 0);
    mha_kernel<<<NB * 8, 32, 0, stream>>>(qkvbuf, o16);
    linear(o16, woT[i], tbuf, nullptr, NIMG, HID, HID, tf[i].bo, nullptr, nullptr, 0, 0);
    ln_kernel<<<NIMG, HID, 0, stream>>>(tbuf, cx32, tf[i].g1, tf[i].b1, ox32, ox16, 0);
    linear(ox16, wf1T[i], nullptr, ffh, NIMG, HID, 512, tf[i].bf1, nullptr, nullptr, 1, 0);
    linear(ffh, wf2T[i], tbuf, nullptr, NIMG, 512, HID, tf[i].bf2, nullptr, nullptr, 0, 0);
    ln_kernel<<<NIMG, HID, 0, stream>>>(tbuf, ox32, tf[i].g2, tf[i].b2, cx32, cx16, 0);
  }
  // ---- 7) encoder MLP -> zp ----
  linear(cx16, encT[0], nullptr, m0h, NB, NJ * HID, 512, enc[0].b, enc[0].g, enc[0].beta, 1, 1);
  linear(m0h, encT[1], nullptr, m1h, NB, 512, 256, enc[1].b, enc[1].g, enc[1].beta, 1, 1);
  linear(m1h, encT[2], zp, nullptr, NB, 256, 40, enc[2].b, nullptr, nullptr, 0, 0);
  // ---- 8) VAE reparameterization ----
  float* outF = (float*)d_out;
  vae_kernel<<<1, 64, 0, stream>>>(zp, d_eps, outF + 76, outF + 116, outF + 156, zws16);
  // ---- 9) decoder + refine ----
  linear(zws16, decT[0], nullptr, m0h, NB, 20, 256, dec[0].b, dec[0].g, dec[0].beta, 1, 1);
  linear(m0h, decT[1], nullptr, m1h, NB, 256, 512, dec[1].b, dec[1].g, dec[1].beta, 1, 1);
  linear(m1h, decT[2], nullptr, hrec16, NB, 512, NJ * HID, dec[2].b, nullptr, nullptr, 0, 0);
  linear(hrec16, refT[0], nullptr, m0h, NB, NJ * HID, 512, refl[0].b, refl[0].g, refl[0].beta, 1, 1);
  linear(m0h, refT[1], nullptr, m1h, NB, 512, 256, refl[1].b, refl[1].g, refl[1].beta, 1, 1);
  linear(m1h, refT[2], outF, nullptr, NB, 256, NJ * 2, refl[2].b, nullptr, nullptr, 0, 0);
}
